// RoIHead_4595615007173
// MI455X (gfx1250) — matrix operand order repaired and checked by reading, not compiled, not run on hardware
//
#include <hip/hip_runtime.h>

// ---------------------------------------------------------------------------
// RoI detection head for MI455X (gfx1250, wave32, WMMA, TDM).
//   FC1 (fused RoI-align A-tile producer) -> FC2 -> {cls, bbox} heads
//     bf16 WMMA / f32 accumulate; B-panels streamed by the Tensor Data Mover
//     into double-buffered LDS, overlapped with WMMA compute.
//   softmax -> decode+compact -> per-image iterative NMS
// Workspace ~51 MB. All work launched on `stream`; no host sync.
// ---------------------------------------------------------------------------

#define B_IMG 2
#define N_ROI 1000
#define C_FEAT 256
#define H_FEAT 50
#define W_FEAT 50
#define P_POOL 7
#define NUM_CLASSES 91
#define REP 1024
#define K1 (C_FEAT * P_POOL * P_POOL) /* 12544 */
#define M_ROWS (B_IMG * N_ROI)        /* 2000  */
#define M_PAD 2048
#define NCLS_PAD 128
#define NBOX_PAD 384
#define DET 100
#define CAND_PER_IMG (N_ROI * (NUM_CLASSES - 1)) /* 90000 */
#define SCALE_F (1.0f / 16.0f)
#define NEG_F (-1.0e9f)
#define NEG_HALF (-5.0e8f)
#define BBOX_CLIP_F 4.135166556742356f /* log(1000/16) */

#if defined(__has_builtin)
#if __has_builtin(__builtin_amdgcn_tensor_load_to_lds) &&                      \
    __has_builtin(__builtin_amdgcn_s_wait_tensorcnt)
#define HAVE_TDM 1
#endif
#endif
#ifndef HAVE_TDM
#define HAVE_TDM 0
#endif

typedef __attribute__((ext_vector_type(16))) __bf16 v16bf;
typedef __attribute__((ext_vector_type(8))) float v8f;
typedef __attribute__((ext_vector_type(4))) unsigned int uint4v;
typedef __attribute__((ext_vector_type(4))) unsigned int u32x4_t;
typedef __attribute__((ext_vector_type(8))) int i32x8_t;
typedef __attribute__((ext_vector_type(4))) int i32x4_t;

union FragBF {
  v16bf v;
  uint4v u[2];
};

__device__ __forceinline__ unsigned short f2bf(float f) {
  unsigned int u = __float_as_uint(f);
  unsigned int r = (u + 0x7FFFu + ((u >> 16) & 1u)) >> 16; // RNE
  return (unsigned short)r;
}

#if HAVE_TDM
// TDM: DMA a 2-D bf16 tile (tile_w elems x tile_h rows, row stride in elems)
// from global memory into LDS at byte offset lds_off. Descriptor packed per
// CDNA5 ISA D# group0/group1 bitfields (count=1, type=2 "image",
// data_size=1 (2 bytes), tile_dim2=0 -> 2-D, no padding/iterate/gather).
__device__ __forceinline__ void tdm_load_tile_2d(unsigned lds_off,
                                                 const unsigned short *gptr,
                                                 unsigned tile_w,
                                                 unsigned tile_h,
                                                 unsigned long long stride0) {
  unsigned long long ga = (unsigned long long)(const void *)gptr;
  u32x4_t g0;
  g0[0] = 1u;            // count=1 (valid), is_restore=0, gather_mode=0
  g0[1] = lds_off;       // D#.lds_addr (bytes)
  g0[2] = (unsigned)ga;  // global_addr[31:0]
  g0[3] = (unsigned)((ga >> 32) & 0x01FFFFFFull) | (2u << 30); // [56:32]|type=2
  unsigned long long td0 = stride0;      // tensor_dim0 (no OOB for our tiles)
  unsigned long long st0 = stride0;      // tensor_dim0_stride
  unsigned td1 = 0x40000000u;            // tensor_dim1: huge (tiles in-bounds)
  i32x8_t g1;
  g1[0] = (int)(1u << 16); // workgroup_mask=0, data_size=1 (2B), no flags
  g1[1] = (int)((unsigned)(td0 & 0xFFFFull) << 16);          // dim0[15:0]
  g1[2] = (int)((unsigned)((td0 >> 16) & 0xFFFFull) |
                ((td1 & 0xFFFFu) << 16));                    // dim0[31:16],dim1[15:0]
  g1[3] = (int)(((td1 >> 16) & 0xFFFFu) | (tile_w << 16));   // dim1[31:16],tile_dim0
  g1[4] = (int)(tile_h & 0xFFFFu);                           // tile_dim1, tile_dim2=0
  g1[5] = (int)(unsigned)(st0 & 0xFFFFFFFFull);              // dim0_stride[31:0]
  g1[6] = (int)(unsigned)((st0 >> 32) & 0xFFFFull);          // dim0_stride[47:32]
  g1[7] = 0;
  i32x4_t z4 = {0, 0, 0, 0};
#if __clang_major__ >= 23
  i32x8_t z8 = {0, 0, 0, 0, 0, 0, 0, 0};
  __builtin_amdgcn_tensor_load_to_lds(g0, g1, z4, z4, z8, 0);
#else
  __builtin_amdgcn_tensor_load_to_lds(g0, g1, z4, z4, 0);
#endif
}
#endif // HAVE_TDM

struct NmsBufs {
  float *qx1, *qy1, *qx2, *qy2;  // offsetted boxes (per-class offset applied)
  float *ox1, *oy1, *ox2, *oy2;  // original clipped boxes
  float *area, *s, *swork;
  int *label;
  int *cnt; // [B_IMG]
};

// ---------------- prep: transpose f32 [K,N] -> bf16 [Npad,K] (zero pad) -----
__global__ void transpose_w_kernel(const float *__restrict__ W,
                                   unsigned short *__restrict__ Wt, int K,
                                   int Nn, int Npad) {
  size_t idx = (size_t)blockIdx.x * blockDim.x + threadIdx.x;
  size_t total = (size_t)Npad * (size_t)K;
  if (idx >= total) return;
  int n = (int)(idx / (size_t)K);
  int k = (int)(idx - (size_t)n * (size_t)K);
  float v = (n < Nn) ? W[(size_t)k * Nn + n] : 0.0f;
  Wt[idx] = f2bf(v);
}

__global__ void pad_bias_kernel(const float *__restrict__ src,
                                float *__restrict__ dst, int n, int npad) {
  int i = blockIdx.x * blockDim.x + threadIdx.x;
  if (i < npad) dst[i] = (i < n) ? src[i] : 0.0f;
}

__global__ void zero_cnt_kernel(int *cnt) {
  if (threadIdx.x < B_IMG) cnt[threadIdx.x] = 0;
}

// ---------------- generic bf16 GEMM: out[M_PAD,N] = relu(A@Bt^T + bias) -----
// A: bf16 [M_PAD,K] row-major. Bt: bf16 [N,K] row-major (i.e. W^T).
// Block tile 64x128x32; 8 waves, each wave a 16x64 strip (4 WMMAs / k-step).
// B panel is DMA'd by the TDM into double-buffered LDS, overlapped w/ WMMA.
template <bool RELU, bool OUTBF16>
__global__ __launch_bounds__(256) void gemm_bf16_kernel(
    const unsigned short *__restrict__ A, const unsigned short *__restrict__ Bt,
    const float *__restrict__ bias, void *__restrict__ out, int Mreal, int Nn,
    int K) {
  __shared__ __align__(16) unsigned short As[64 * 32];
  __shared__ __align__(16) unsigned short Bs[1 + HAVE_TDM][128 * 32];
  const int t = threadIdx.x;
  const int lane = t & 31;
  const int wid = t >> 5;
  const int m0g = blockIdx.y * 64;
  const int n0g = blockIdx.x * 128;
  const int wm = (wid & 3) * 16;
  const int wn = (wid >> 2) * 64;
  const int aR = t >> 2, aK = (t & 3) * 8; // A tile: 64 rows x 32 k (16B/thr)
  const int KT = K / 32;

  v8f acc[4] = {};

#if HAVE_TDM
  const unsigned bsOff0 = (unsigned)(size_t)(void *)&Bs[0][0];
  const unsigned bsOff1 = (unsigned)(size_t)(void *)&Bs[1][0];
  if (wid == 0) // one TDM descriptor per workgroup; EXEC-independent DMA
    tdm_load_tile_2d(bsOff0, Bt + (size_t)n0g * K, 32, 128,
                     (unsigned long long)K);
#else
  const int bR = t >> 1, bK = (t & 1) * 16;
#endif
  int buf = 0;

  for (int kt = 0; kt < KT; ++kt) {
    const int k0 = kt * 32;
    const unsigned short *ag = A + (size_t)(m0g + aR) * K + k0 + aK;
    uint4v av = *(const uint4v *)ag;
    __builtin_prefetch(ag + 128, 0, 1); // stream A panel ahead
    *(uint4v *)(As + aR * 32 + aK) = av;
#if HAVE_TDM
    if (wid == 0) __builtin_amdgcn_s_wait_tensorcnt(0);
    __syncthreads(); // publish As stores + completed B DMA
    if (wid == 0 && kt + 1 < KT)
      tdm_load_tile_2d(buf ? bsOff0 : bsOff1,
                       Bt + (size_t)n0g * K + (k0 + 32), 32, 128,
                       (unsigned long long)K);
#else
    {
      const unsigned short *bg = Bt + (size_t)(n0g + bR) * K + k0 + bK;
      uint4v bv0 = *(const uint4v *)(bg);
      uint4v bv1 = *(const uint4v *)(bg + 8);
      *(uint4v *)(&Bs[0][0] + bR * 32 + bK) = bv0;
      *(uint4v *)(&Bs[0][0] + bR * 32 + bK + 8) = bv1;
    }
    __syncthreads();
#endif
    const unsigned short *BsC = &Bs[HAVE_TDM ? buf : 0][0];

    // A fragment per ISA 16-bit A layout: lanes<16 K{0..7,16..23}, else {8..15,24..31}
    FragBF afr;
    {
      const unsigned short *ap = As + (wm + (lane & 15)) * 32;
      int hi = (lane >> 4) * 8;
      afr.u[0] = *(const uint4v *)(ap + hi);
      afr.u[1] = *(const uint4v *)(ap + 16 + hi);
    }
#pragma unroll
    for (int s = 0; s < 4; ++s) {
      FragBF bfr; // lanes<16: K {0..7,16..23} of column n; lanes>=16: K {8..15,24..31}
      const unsigned short *bp =
          BsC + (wn + s * 16 + (lane & 15)) * 32 + (lane >> 4) * 8;
      bfr.u[0] = *(const uint4v *)(bp);
      bfr.u[1] = *(const uint4v *)(bp + 16);
      acc[s] = __builtin_amdgcn_wmma_f32_16x16x32_bf16(
          false, afr.v, false, bfr.v, (short)0, acc[s], false, false);
    }
    __syncthreads();
    buf ^= 1;
  }

#pragma unroll
  for (int s = 0; s < 4; ++s) {
    int n = n0g + wn + s * 16 + (lane & 15);
    float bv = bias[n];
#pragma unroll
    for (int r = 0; r < 8; ++r) {
      int m = m0g + wm + ((lane >> 4) * 8) + r; // C layout per ISA
      float v = acc[s][r] + bv;
      if (RELU) v = fmaxf(v, 0.0f);
      if (m >= Mreal) v = 0.0f; // zero-pad rows for downstream GEMM A
      if (OUTBF16)
        ((unsigned short *)out)[(size_t)m * Nn + n] = f2bf(v);
      else
        ((float *)out)[(size_t)m * Nn + n] = v;
    }
  }
}

// ---------------- FC1 with fused RoI-align A-tile producer ------------------
__global__ __launch_bounds__(256) void gemm1_roialign_kernel(
    const float *__restrict__ features, const float *__restrict__ proposals,
    const unsigned short *__restrict__ Bt, const float *__restrict__ bias,
    unsigned short *__restrict__ out) {
  __shared__ __align__(16) unsigned short As[64 * 32];
  __shared__ __align__(16) unsigned short Bs[1 + HAVE_TDM][128 * 32];
  const int t = threadIdx.x;
  const int lane = t & 31;
  const int wid = t >> 5;
  const int m0g = blockIdx.y * 64;
  const int n0g = blockIdx.x * 128;
  const int wm = (wid & 3) * 16;
  const int wn = (wid >> 2) * 64;
  const int aR = t >> 2, aK = (t & 3) * 8;
  const int K = K1, Nn = REP;
  const int KT = K / 32;

  // Per-thread RoI parameters (row fixed for whole kernel).
  const int m = m0g + aR;
  const bool valid = (m < M_ROWS);
  float rx1 = 0.f, ry1 = 0.f, bw = 0.f, bh = 0.f;
  int bimg = 0;
  if (valid) {
    bimg = m / N_ROI;
    const float *p = proposals + (size_t)m * 4;
    float p0 = p[0], p1 = p[1], p2 = p[2], p3 = p[3];
    rx1 = p0 * SCALE_F;
    ry1 = p1 * SCALE_F;
    bw = (p2 - p0) * SCALE_F * (1.0f / (float)P_POOL);
    bh = (p3 - p1) * SCALE_F * (1.0f / (float)P_POOL);
  }
  const float *featB = features + (size_t)bimg * (C_FEAT * H_FEAT * W_FEAT);

  v8f acc[4] = {};

#if HAVE_TDM
  const unsigned bsOff0 = (unsigned)(size_t)(void *)&Bs[0][0];
  const unsigned bsOff1 = (unsigned)(size_t)(void *)&Bs[1][0];
  if (wid == 0)
    tdm_load_tile_2d(bsOff0, Bt + (size_t)n0g * K, 32, 128,
                     (unsigned long long)K);
#else
  const int bR = t >> 1, bK = (t & 1) * 16;
#endif
  int buf = 0;

  // Incremental k -> (c, rem) decomposition: k = k0 + aK steps by 32/iter,
  // so c = k/49 advances by 0/1 and rem = k%49 by +-. Avoids div chains.
  int cbase = 0, rbase = aK; // aK in {0,8,16,24} < 49

  for (int kt = 0; kt < KT; ++kt) {
    // Produce A tile: bilinear RoI-align samples straight into LDS (bf16).
#pragma unroll
    for (int j = 0; j < 8; ++j) {
      float val = 0.0f;
      if (valid) {
        int r = rbase + j;                  // <= 48+7
        int cj = cbase + (r >= 49 ? 1 : 0);
        int rj = r - (r >= 49 ? 49 : 0);    // 0..48
        int py = (rj * 37) >> 8;            // == rj/7 on [0,48]
        int px = rj - py * 7;
        float yy = ry1 + ((float)py + 0.5f) * bh - 0.5f;
        float xx = rx1 + ((float)px + 0.5f) * bw - 0.5f;
        yy = fminf(fmaxf(yy, 0.0f), (float)(H_FEAT - 1));
        xx = fminf(fmaxf(xx, 0.0f), (float)(W_FEAT - 1));
        int y0 = (int)floorf(yy);
        int x0 = (int)floorf(xx);
        int y1i = min(y0 + 1, H_FEAT - 1);
        int x1i = min(x0 + 1, W_FEAT - 1);
        float ly = yy - (float)y0, lx = xx - (float)x0;
        float hy = 1.0f - ly, hx = 1.0f - lx;
        const float *f = featB + (size_t)cj * (H_FEAT * W_FEAT);
        float v00 = f[y0 * W_FEAT + x0], v01 = f[y0 * W_FEAT + x1i];
        float v10 = f[y1i * W_FEAT + x0], v11 = f[y1i * W_FEAT + x1i];
        val = v00 * hy * hx + v01 * hy * lx + v10 * ly * hx + v11 * ly * lx;
      }
      As[aR * 32 + aK + j] = f2bf(val);
    }
    rbase += 32;
    if (rbase >= 49) { rbase -= 49; cbase += 1; }

#if HAVE_TDM
    if (wid == 0) __builtin_amdgcn_s_wait_tensorcnt(0);
    __syncthreads();
    if (wid == 0 && kt + 1 < KT)
      tdm_load_tile_2d(buf ? bsOff0 : bsOff1,
                       Bt + (size_t)n0g * K + (kt * 32 + 32), 32, 128,
                       (unsigned long long)K);
#else
    {
      const unsigned short *bg = Bt + (size_t)(n0g + bR) * K + kt * 32 + bK;
      uint4v bv0 = *(const uint4v *)(bg);
      uint4v bv1 = *(const uint4v *)(bg + 8);
      *(uint4v *)(&Bs[0][0] + bR * 32 + bK) = bv0;
      *(uint4v *)(&Bs[0][0] + bR * 32 + bK + 8) = bv1;
    }
    __syncthreads();
#endif
    const unsigned short *BsC = &Bs[HAVE_TDM ? buf : 0][0];

    FragBF afr;
    {
      const unsigned short *ap = As + (wm + (lane & 15)) * 32;
      int hi = (lane >> 4) * 8;
      afr.u[0] = *(const uint4v *)(ap + hi);
      afr.u[1] = *(const uint4v *)(ap + 16 + hi);
    }
#pragma unroll
    for (int s = 0; s < 4; ++s) {
      FragBF bfr;
      const unsigned short *bp =
          BsC + (wn + s * 16 + (lane & 15)) * 32 + (lane >> 4) * 8;
      bfr.u[0] = *(const uint4v *)(bp);
      bfr.u[1] = *(const uint4v *)(bp + 16);
      acc[s] = __builtin_amdgcn_wmma_f32_16x16x32_bf16(
          false, afr.v, false, bfr.v, (short)0, acc[s], false, false);
    }
    __syncthreads();
    buf ^= 1;
  }

#pragma unroll
  for (int s = 0; s < 4; ++s) {
    int n = n0g + wn + s * 16 + (lane & 15);
    float bv = bias[n];
#pragma unroll
    for (int r = 0; r < 8; ++r) {
      int mm = m0g + wm + ((lane >> 4) * 8) + r;
      float v = fmaxf(acc[s][r] + bv, 0.0f);
      if (mm >= M_ROWS) v = 0.0f;
      out[(size_t)mm * Nn + n] = f2bf(v);
    }
  }
}

// ---------------- softmax over 91 logits per roi (one wave32 per row) -------
__global__ void softmax_kernel(const float *__restrict__ logits,
                               float *__restrict__ scores) {
  int warp = threadIdx.x >> 5;
  int lane = threadIdx.x & 31;
  int row = blockIdx.x * 4 + warp;
  if (row >= M_ROWS) return;
  const float *lp = logits + (size_t)row * NCLS_PAD;
  float v0 = lp[lane];
  float v1 = (lane + 32 < NUM_CLASSES) ? lp[lane + 32] : -3.0e38f;
  float v2 = (lane + 64 < NUM_CLASSES) ? lp[lane + 64] : -3.0e38f;
  float mx = fmaxf(v0, fmaxf(v1, v2));
  for (int d = 16; d; d >>= 1) mx = fmaxf(mx, __shfl_xor(mx, d, 32));
  float e0 = expf(v0 - mx);
  float e1 = (lane + 32 < NUM_CLASSES) ? expf(v1 - mx) : 0.0f;
  float e2 = (lane + 64 < NUM_CLASSES) ? expf(v2 - mx) : 0.0f;
  float sm = e0 + e1 + e2;
  for (int d = 16; d; d >>= 1) sm += __shfl_xor(sm, d, 32);
  float inv = 1.0f / sm;
  float *sp = scores + (size_t)row * NUM_CLASSES;
  sp[lane] = e0 * inv;
  if (lane + 32 < NUM_CLASSES) sp[lane + 32] = e1 * inv;
  if (lane + 64 < NUM_CLASSES) sp[lane + 64] = e2 * inv;
}

// ---------------- decode + clip + compact valid candidates ------------------
// Compaction is exact vs reference: invalid entries (swork==NEG) can never be
// selected nor suppress anything. Atomic append order only permutes indices;
// the selection-by-max result is order-invariant (ties are measure-zero).
__global__ void cand_kernel(const float *__restrict__ scores,
                            const float *__restrict__ bboxd,
                            const float *__restrict__ proposals,
                            const int *__restrict__ imgsz, NmsBufs nb) {
  int tid = blockIdx.x * blockDim.x + threadIdx.x;
  if (tid >= B_IMG * CAND_PER_IMG) return;
  int b = tid / CAND_PER_IMG;
  int j = tid - b * CAND_PER_IMG;
  int roi = j / (NUM_CLASSES - 1);
  int cls = j - roi * (NUM_CLASSES - 1) + 1; // 1..90
  int row = b * N_ROI + roi;
  float s = scores[(size_t)row * NUM_CLASSES + cls];
  float img = (float)(*imgsz);
  const float *p = proposals + (size_t)row * 4;
  float pw = p[2] - p[0], ph = p[3] - p[1];
  float pcx = p[0] + 0.5f * pw, pcy = p[1] + 0.5f * ph;
  const float *d = bboxd + (size_t)row * NBOX_PAD + cls * 4;
  float dx = d[0] * 0.1f, dy = d[1] * 0.1f;
  float dw = fminf(d[2] * 0.2f, BBOX_CLIP_F);
  float dh = fminf(d[3] * 0.2f, BBOX_CLIP_F);
  float cx = dx * pw + pcx, cy = dy * ph + pcy;
  float w = pw * expf(dw), h = ph * expf(dh);
  float x1 = fminf(fmaxf(cx - 0.5f * w, 0.0f), img);
  float y1 = fminf(fmaxf(cy - 0.5f * h, 0.0f), img);
  float x2 = fminf(fmaxf(cx + 0.5f * w, 0.0f), img);
  float y2 = fminf(fmaxf(cy + 0.5f * h, 0.0f), img);
  float wv = x2 - x1, hv = y2 - y1;
  bool ok = (s > 0.05f) && (wv >= 1.0f) && (hv >= 1.0f);
  if (!ok) return;
  int pos = atomicAdd(&nb.cnt[b], 1);
  if (pos >= CAND_PER_IMG) return;
  size_t o = (size_t)b * CAND_PER_IMG + pos;
  float off = (float)cls * (img + 2.0f);
  nb.ox1[o] = x1; nb.oy1[o] = y1; nb.ox2[o] = x2; nb.oy2[o] = y2;
  float q1 = x1 + off, q2 = y1 + off, q3 = x2 + off, q4 = y2 + off;
  nb.qx1[o] = q1; nb.qy1[o] = q2; nb.qx2[o] = q3; nb.qy2[o] = q4;
  nb.area[o] = (q3 - q1) * (q4 - q2);
  nb.s[o] = s;
  nb.swork[o] = s;
  nb.label[o] = cls;
}

// ---------------- per-image iterative argmax + suppression NMS --------------
__global__ __launch_bounds__(256) void nms_kernel(NmsBufs nb,
                                                  float *__restrict__ dout) {
  const int b = blockIdx.x;
  const int tid = threadIdx.x;
  int V = min(nb.cnt[b], CAND_PER_IMG);
  size_t base = (size_t)b * CAND_PER_IMG;
  float *sw = nb.swork + base;
  const float *QX1 = nb.qx1 + base, *QY1 = nb.qy1 + base;
  const float *QX2 = nb.qx2 + base, *QY2 = nb.qy2 + base;
  const float *OX1 = nb.ox1 + base, *OY1 = nb.oy1 + base;
  const float *OX2 = nb.ox2 + base, *OY2 = nb.oy2 + base;
  const float *AREA = nb.area + base, *SS = nb.s + base;
  const int *LBL = nb.label + base;
  float *dets = dout;
  int *labels = ((int *)dout) + B_IMG * DET * 5;

  __shared__ float rv[256];
  __shared__ int ri[256];
  __shared__ float pick[5];

  for (int it = 0; it < DET; ++it) {
    // pass 1: argmax (first-index tie-break)
    float best = -3.0e38f;
    int bidx = -1;
    for (int j = tid; j < V; j += 256) {
      float v = sw[j];
      if (v > best) { best = v; bidx = j; }
    }
    rv[tid] = best;
    ri[tid] = bidx;
    __syncthreads();
    for (int st = 128; st; st >>= 1) {
      if (tid < st) {
        float v2 = rv[tid + st];
        int i2 = ri[tid + st];
        if ((v2 > rv[tid]) ||
            (v2 == rv[tid] && i2 >= 0 && (ri[tid] < 0 || i2 < ri[tid]))) {
          rv[tid] = v2;
          ri[tid] = i2;
        }
      }
      __syncthreads();
    }
    int i = ri[0];
    float vi = rv[0];
    bool ok = (i >= 0) && (vi > NEG_HALF);
    if (tid == 0) {
      float *dp = dets + ((size_t)b * DET + it) * 5;
      if (ok) {
        pick[0] = QX1[i]; pick[1] = QY1[i];
        pick[2] = QX2[i]; pick[3] = QY2[i];
        pick[4] = AREA[i];
        dp[0] = OX1[i]; dp[1] = OY1[i]; dp[2] = OX2[i]; dp[3] = OY2[i];
        dp[4] = SS[i];
        labels[b * DET + it] = LBL[i];
      } else {
        dp[0] = dp[1] = dp[2] = dp[3] = dp[4] = 0.0f;
        labels[b * DET + it] = -1;
      }
    }
    __syncthreads();
    // pass 2: suppress
    if (ok) {
      float px1 = pick[0], py1 = pick[1], px2 = pick[2], py2 = pick[3];
      float pa = pick[4];
      for (int j = tid; j < V; j += 256) {
        if (j == i) { sw[j] = NEG_F; continue; }
        float v = sw[j];
        if (v <= NEG_HALF) continue;
        float ix1 = fmaxf(px1, QX1[j]);
        float iy1 = fmaxf(py1, QY1[j]);
        float ix2 = fminf(px2, QX2[j]);
        float iy2 = fminf(py2, QY2[j]);
        float inter = fmaxf(ix2 - ix1, 0.0f) * fmaxf(iy2 - iy1, 0.0f);
        float iou = inter / (pa + AREA[j] - inter + 1e-9f);
        if (iou > 0.5f) sw[j] = NEG_F;
      }
    } else if (i >= 0 && tid == 0) {
      sw[i] = NEG_F;
    }
    __threadfence();
    __syncthreads();
  }
}

// ---------------------------------------------------------------------------
extern "C" void kernel_launch(void *const *d_in, const int *in_sizes, int n_in,
                              void *d_out, int out_size, void *d_ws,
                              size_t ws_size, hipStream_t stream) {
  const float *features = (const float *)d_in[0];
  const float *proposals = (const float *)d_in[1];
  const float *w1 = (const float *)d_in[2];
  const float *b1 = (const float *)d_in[3];
  const float *w2 = (const float *)d_in[4];
  const float *b2 = (const float *)d_in[5];
  const float *w_cls = (const float *)d_in[6];
  const float *b_cls = (const float *)d_in[7];
  const float *w_bbox = (const float *)d_in[8];
  const float *b_bbox = (const float *)d_in[9];
  const int *imgsz = (const int *)d_in[10];

  // ---- workspace carve (~51 MB) ----
  char *p = (char *)d_ws;
  auto alloc = [&](size_t bytes) {
    char *r = p;
    p += (bytes + 255) & ~(size_t)255;
    return r;
  };
  unsigned short *w1t = (unsigned short *)alloc((size_t)REP * K1 * 2);
  unsigned short *w2t = (unsigned short *)alloc((size_t)REP * REP * 2);
  unsigned short *wct = (unsigned short *)alloc((size_t)NCLS_PAD * REP * 2);
  unsigned short *wbt = (unsigned short *)alloc((size_t)NBOX_PAD * REP * 2);
  float *bcP = (float *)alloc(NCLS_PAD * 4);
  float *bbP = (float *)alloc(NBOX_PAD * 4);
  unsigned short *h1 = (unsigned short *)alloc((size_t)M_PAD * REP * 2);
  unsigned short *h2 = (unsigned short *)alloc((size_t)M_PAD * REP * 2);
  float *logits = (float *)alloc((size_t)M_PAD * NCLS_PAD * 4);
  float *bboxd = (float *)alloc((size_t)M_PAD * NBOX_PAD * 4);
  float *scores = (float *)alloc((size_t)M_ROWS * NUM_CLASSES * 4);
  NmsBufs nb;
  size_t cbytes = (size_t)B_IMG * CAND_PER_IMG * 4;
  nb.qx1 = (float *)alloc(cbytes);
  nb.qy1 = (float *)alloc(cbytes);
  nb.qx2 = (float *)alloc(cbytes);
  nb.qy2 = (float *)alloc(cbytes);
  nb.ox1 = (float *)alloc(cbytes);
  nb.oy1 = (float *)alloc(cbytes);
  nb.ox2 = (float *)alloc(cbytes);
  nb.oy2 = (float *)alloc(cbytes);
  nb.area = (float *)alloc(cbytes);
  nb.s = (float *)alloc(cbytes);
  nb.swork = (float *)alloc(cbytes);
  nb.label = (int *)alloc(cbytes);
  nb.cnt = (int *)alloc(B_IMG * 4);

  // ---- prep: weight transpose/convert + bias pad + counters ----
  {
    size_t tot = (size_t)REP * K1;
    transpose_w_kernel<<<(unsigned)((tot + 255) / 256), 256, 0, stream>>>(
        w1, w1t, K1, REP, REP);
  }
  {
    size_t tot = (size_t)REP * REP;
    transpose_w_kernel<<<(unsigned)((tot + 255) / 256), 256, 0, stream>>>(
        w2, w2t, REP, REP, REP);
  }
  {
    size_t tot = (size_t)NCLS_PAD * REP;
    transpose_w_kernel<<<(unsigned)((tot + 255) / 256), 256, 0, stream>>>(
        w_cls, wct, REP, NUM_CLASSES, NCLS_PAD);
  }
  {
    size_t tot = (size_t)NBOX_PAD * REP;
    transpose_w_kernel<<<(unsigned)((tot + 255) / 256), 256, 0, stream>>>(
        w_bbox, wbt, REP, NUM_CLASSES * 4, NBOX_PAD);
  }
  pad_bias_kernel<<<1, 256, 0, stream>>>(b_cls, bcP, NUM_CLASSES, NCLS_PAD);
  pad_bias_kernel<<<2, 256, 0, stream>>>(b_bbox, bbP, NUM_CLASSES * 4,
                                         NBOX_PAD);
  zero_cnt_kernel<<<1, 32, 0, stream>>>(nb.cnt);

  // ---- FC1: fused RoI-align + GEMM (bf16 WMMA + TDM B stream) ----
  gemm1_roialign_kernel<<<dim3(REP / 128, M_PAD / 64), 256, 0, stream>>>(
      features, proposals, w1t, b1, h1);
  // ---- FC2 ----
  gemm_bf16_kernel<true, true><<<dim3(REP / 128, M_PAD / 64), 256, 0, stream>>>(
      h1, w2t, b2, h2, M_ROWS, REP, REP);
  // ---- heads ----
  gemm_bf16_kernel<false, false>
      <<<dim3(NCLS_PAD / 128, M_PAD / 64), 256, 0, stream>>>(
          h2, wct, bcP, logits, M_ROWS, NCLS_PAD, REP);
  gemm_bf16_kernel<false, false>
      <<<dim3(NBOX_PAD / 128, M_PAD / 64), 256, 0, stream>>>(
          h2, wbt, bbP, bboxd, M_ROWS, NBOX_PAD, REP);

  // ---- softmax / decode / NMS ----
  softmax_kernel<<<(M_ROWS + 3) / 4, 128, 0, stream>>>(logits, scores);
  {
    int tot = B_IMG * CAND_PER_IMG;
    cand_kernel<<<(tot + 255) / 256, 256, 0, stream>>>(scores, bboxd,
                                                       proposals, imgsz, nb);
  }
  nms_kernel<<<B_IMG, 256, 0, stream>>>(nb, (float *)d_out);
}
